// FinCastMini_81905026334893
// MI455X (gfx1250) — matrix operand order repaired and checked by reading, not compiled, not run on hardware
//
#include <hip/hip_runtime.h>
#include <hip/hip_bf16.h>

typedef _Float16 half8 __attribute__((ext_vector_type(8)));
typedef _Float16 v16h  __attribute__((ext_vector_type(16)));
typedef float    v8f   __attribute__((ext_vector_type(8)));

// Problem constants (from reference)
static constexpr int Bc  = 8;
static constexpr int Sc  = 1024;
static constexpr int Fc  = 8;
static constexpr int Dc  = 512;
static constexpr int Hc  = 8;
static constexpr int Lc  = 3;
static constexpr int Ec  = 4;
static constexpr int HDc = 64;
static constexpr int Tc  = Bc * Sc;           // 8192 tokens

// ---------------------------------------------------------------------------
// Batched WMMA GEMM:  C[M,N] = A[M,K] @ W[N,K]^T + bias, optional ReLU,
// optional f32 and/or f16 outputs with independent ld / batch strides.
//
// Block = 256 threads = 8 waves arranged 4(M) x 2(N); wave tile 32x64 =
// 2x4 v_wmma_f32_16x16x32_f16 accumulators (8 WMMA / wave / K-step).
// A tile (128x32 f16) is double-buffered in LDS (one barrier per K-step,
// next tile prefetched into registers during compute). B fragments are
// two 16B-per-lane reads (32B apart) straight from global: all weights are
// L2-resident on MI455X's 192MB L2, so B reuse costs L2 bandwidth, not HBM.
// Requires: M % 128 == 0, K % 32 == 0, K >= 64 (true for all call sites).
// ---------------------------------------------------------------------------
__global__ __launch_bounds__(256) void gemm_wmma_f16(
    const _Float16* __restrict__ A, long long sA,
    const _Float16* __restrict__ W, long long sW,
    const float* __restrict__ bias,
    float* __restrict__ C, int ldC, long long sC,
    _Float16* __restrict__ C16, int ldC16, long long sC16,
    int M, int N, int K, int relu)
{
    constexpr int BM = 128, BN = 128, BK = 32;
    constexpr int AST = 48;                         // padded LDS row stride (halves)
    __shared__ __align__(16) _Float16 As[2][BM * AST];

    const int z = blockIdx.z;
    A += (long long)z * sA;
    W += (long long)z * sW;

    const int tid  = threadIdx.x;
    const int lane = tid & 31;
    const int wave = tid >> 5;
    const int wn   = wave & 1;          // 2 waves along N (64 cols each)
    const int wm   = wave >> 1;         // 4 waves along M (32 rows each)
    const int row0 = blockIdx.y * BM;
    const int col0 = blockIdx.x * BN;
    const int lhi  = lane >> 4;         // 0 or 1
    const int llo  = lane & 15;

    // staging assignment: 128 rows x 32 halves = 512 8-half chunks, 2/thread
    const int r0s  = tid >> 2;                  // chunk tid
    const int seg0 = (tid & 3) << 3;
    const int r1s  = (tid + 256) >> 2;          // chunk tid+256
    const int seg1 = ((tid + 256) & 3) << 3;

    v8f acc[2][4];
    {
        v8f zero = {};
#pragma unroll
        for (int mt = 0; mt < 2; ++mt)
#pragma unroll
            for (int nt = 0; nt < 4; ++nt) acc[mt][nt] = zero;
    }

    // prologue: stage K-tile 0 into buffer 0
    {
        const half8 v0 = *(const half8*)(A + (long long)(row0 + r0s) * K + seg0);
        const half8 v1 = *(const half8*)(A + (long long)(row0 + r1s) * K + seg1);
        *(half8*)(&As[0][r0s * AST + seg0]) = v0;
        *(half8*)(&As[0][r1s * AST + seg1]) = v1;
    }

    const int ktiles = K / BK;
    for (int kt = 0; kt < ktiles; ++kt) {
        const int k0 = kt * BK;
        __syncthreads();                        // staging of buffer (kt&1) done

        // prefetch next A tile into registers while we compute
        const bool more = (kt + 1 < ktiles);
        half8 n0 = {}, n1 = {};
        if (more) {
            n0 = *(const half8*)(A + (long long)(row0 + r0s) * K + k0 + BK + seg0);
            n1 = *(const half8*)(A + (long long)(row0 + r1s) * K + k0 + BK + seg1);
        }

        const _Float16* Sb = As[kt & 1];

        // A fragments from LDS (ISA 16-bit A layout: lanes 0-15 K=0..7/16..23,
        // lanes 16-31 K=8..15/24..31, M = lane&15)
        v16h afrag[2];
#pragma unroll
        for (int mt = 0; mt < 2; ++mt) {
            const int ml = wm * 32 + mt * 16 + llo;
            const _Float16* p = &Sb[ml * AST + (lhi << 3)];
            half8 a0 = *(const half8*)p;
            half8 a1 = *(const half8*)(p + 16);
            afrag[mt] = __builtin_shufflevector(a0, a1,
                0,1,2,3,4,5,6,7,8,9,10,11,12,13,14,15);
        }

        // B fragments direct from global (row n of W contiguous in K;
        // lanes 0-15 hold K=0..7/16..23, lanes 16-31 K=8..15/24..31)
        v16h bfrag[4];
#pragma unroll
        for (int nt = 0; nt < 4; ++nt) {
            int n = col0 + wn * 64 + nt * 16 + llo;
            if (n >= N) n = N - 1;                      // clamp (store guarded)
            const _Float16* p = W + (long long)n * K + k0 + (lhi << 3);
            half8 b0 = *(const half8*)p;
            half8 b1 = *(const half8*)(p + 16);
            bfrag[nt] = __builtin_shufflevector(b0, b1,
                0,1,2,3,4,5,6,7,8,9,10,11,12,13,14,15);
            if (more)
                __builtin_prefetch((const void*)(p + BK), 0, 0);
        }

#pragma unroll
        for (int mt = 0; mt < 2; ++mt)
#pragma unroll
            for (int nt = 0; nt < 4; ++nt)
                acc[mt][nt] = __builtin_amdgcn_wmma_f32_16x16x32_f16(
                    false, afrag[mt], false, bfrag[nt],
                    (short)0, acc[mt][nt], false, false);

        // stage next tile into the other buffer; safe without a second
        // barrier because that buffer is only read after the next barrier
        if (more) {
            _Float16* Db = As[(kt + 1) & 1];
            *(half8*)(&Db[r0s * AST + seg0]) = n0;
            *(half8*)(&Db[r1s * AST + seg1]) = n1;
        }
    }

    // Epilogue: D layout = vgpr j -> row j + 8*(lane>=16); col = lane&15
#pragma unroll
    for (int mt = 0; mt < 2; ++mt) {
#pragma unroll
        for (int nt = 0; nt < 4; ++nt) {
            const int col = col0 + wn * 64 + nt * 16 + llo;
            if (col >= N) continue;
            const float bv = bias ? bias[col] : 0.0f;
#pragma unroll
            for (int j = 0; j < 8; ++j) {
                const int row = row0 + wm * 32 + mt * 16 + (lhi << 3) + j;
                if (row >= M) continue;
                float v = acc[mt][nt][j] + bv;
                if (relu) v = fmaxf(v, 0.0f);
                if (C)   C[(long long)z * sC   + (long long)row * ldC   + col] = v;
                if (C16) C16[(long long)z * sC16 + (long long)row * ldC16 + col] = (_Float16)v;
            }
        }
    }
}

// ---------------------------------------------------------------------------
// Instance-norm stats over time axis (per (b,f), ddof=1 like torch std)
// ---------------------------------------------------------------------------
__global__ void inst_stats(const float* __restrict__ x,
                           float* __restrict__ mean, float* __restrict__ istd)
{
    __shared__ float rs[256], rss[256];
    const int bf = blockIdx.x;
    const int b = bf >> 3, f = bf & 7;
    const float* p = x + (long long)b * Sc * Fc + f;
    float s = 0.f, ss = 0.f;
    for (int i = threadIdx.x; i < Sc; i += 256) {
        float v = p[(long long)i * Fc];
        s += v; ss += v * v;
    }
    rs[threadIdx.x] = s; rss[threadIdx.x] = ss; __syncthreads();
    for (int s2 = 128; s2 > 0; s2 >>= 1) {
        if (threadIdx.x < s2) { rs[threadIdx.x] += rs[threadIdx.x + s2];
                                rss[threadIdx.x] += rss[threadIdx.x + s2]; }
        __syncthreads();
    }
    if (threadIdx.x == 0) {
        float m = rs[0] / (float)Sc;
        float var = (rss[0] - (float)Sc * m * m) / (float)(Sc - 1);
        mean[bf] = m;
        istd[bf] = 1.0f / (sqrtf(fmaxf(var, 0.0f)) + 1e-5f);
    }
}

// ---------------------------------------------------------------------------
// h = xn @ Wp^T + bp + femb[freq] + pos_encoding  (K=8 -> plain VALU)
// ---------------------------------------------------------------------------
__global__ void proj_embed(const float* __restrict__ x,
                           const float* __restrict__ mean, const float* __restrict__ istd,
                           const float* __restrict__ Wp, const float* __restrict__ bp,
                           const float* __restrict__ femb, const int* __restrict__ freq,
                           float* __restrict__ h32, _Float16* __restrict__ h16)
{
    const long long i = (long long)blockIdx.x * 256 + threadIdx.x;   // < T*D
    const int t = (int)(i >> 9);
    const int d = (int)(i & 511);
    const int b = t >> 10, s = t & 1023;

    float acc = bp[d] + femb[freq[0] * Dc + d];
    const int i2 = d & ~1;
    const float div = __expf((float)i2 * (-0.0179889460f));  // -ln(10000)/512
    const float ang = (float)s * div;
    acc += (d & 1) ? __cosf(ang) : __sinf(ang);

    const float* xr = x + (long long)t * Fc;
    const float* mb = mean + b * Fc;
    const float* ib = istd + b * Fc;
#pragma unroll
    for (int f = 0; f < Fc; ++f)
        acc += (xr[f] - mb[f]) * ib[f] * Wp[d * Fc + f];
    h32[i] = acc;
    h16[i] = (_Float16)acc;
}

// ---------------------------------------------------------------------------
// Split QKV [T,1536] f16 into q16/k16 [B,H,S,HD] and transposed v16t [B,H,HD,S]
// ---------------------------------------------------------------------------
__global__ void split_qkv(const _Float16* __restrict__ qkv,
                          _Float16* __restrict__ q, _Float16* __restrict__ k,
                          _Float16* __restrict__ vt)
{
    const long long i = (long long)blockIdx.x * 256 + threadIdx.x;   // < B*S*D
    const int t = (int)(i >> 9);
    const int d = (int)(i & 511);
    const int b = t >> 10, s = t & 1023;
    const int hh = d >> 6, hd = d & 63;
    const long long qi = (((long long)b * Hc + hh) * Sc + s) * HDc + hd;
    const _Float16* row = qkv + (long long)t * (3 * Dc);
    q[qi] = row[d];
    k[qi] = row[Dc + d];
    vt[(((long long)b * Hc + hh) * HDc + hd) * Sc + s] = row[2 * Dc + d];
}

// ---------------------------------------------------------------------------
// Row softmax with scale, f32 scores -> f16 probabilities (cols = 1024)
// ---------------------------------------------------------------------------
__global__ void softmax_rows(const float* __restrict__ S32, long long sB,
                             _Float16* __restrict__ P, long long sBP, float scale)
{
    __shared__ float red[256];
    const int row = blockIdx.x, z = blockIdx.z, tid = threadIdx.x;
    const float* rp = S32 + (long long)z * sB + (long long)row * Sc;
    _Float16*   op  = P   + (long long)z * sBP + (long long)row * Sc;
    float v[4];
#pragma unroll
    for (int i = 0; i < 4; ++i) v[i] = rp[tid + i * 256] * scale;
    float m = fmaxf(fmaxf(v[0], v[1]), fmaxf(v[2], v[3]));
    red[tid] = m; __syncthreads();
    for (int s2 = 128; s2 > 0; s2 >>= 1) {
        if (tid < s2) red[tid] = fmaxf(red[tid], red[tid + s2]);
        __syncthreads();
    }
    m = red[0]; __syncthreads();
    float sum = 0.f;
#pragma unroll
    for (int i = 0; i < 4; ++i) { v[i] = __expf(v[i] - m); sum += v[i]; }
    red[tid] = sum; __syncthreads();
    for (int s2 = 128; s2 > 0; s2 >>= 1) {
        if (tid < s2) red[tid] += red[tid + s2];
        __syncthreads();
    }
    const float inv = 1.0f / red[0];
#pragma unroll
    for (int i = 0; i < 4; ++i) op[tid + i * 256] = (_Float16)(v[i] * inv);
}

// ---------------------------------------------------------------------------
// h = LayerNorm(h + r) * g + beta ; also refresh f16 copy. One block per token.
// ---------------------------------------------------------------------------
__global__ void add_ln(float* __restrict__ h, const float* __restrict__ r,
                       const float* __restrict__ g, const float* __restrict__ be,
                       _Float16* __restrict__ h16)
{
    __shared__ float red[256];
    const int t = blockIdx.x, tid = threadIdx.x;
    const long long base = (long long)t * Dc;
    float v0 = h[base + tid]        + r[base + tid];
    float v1 = h[base + 256 + tid]  + r[base + 256 + tid];
    red[tid] = v0 + v1; __syncthreads();
    for (int s2 = 128; s2 > 0; s2 >>= 1) {
        if (tid < s2) red[tid] += red[tid + s2];
        __syncthreads();
    }
    const float m = red[0] * (1.0f / (float)Dc);
    __syncthreads();
    const float d0 = v0 - m, d1 = v1 - m;
    red[tid] = d0 * d0 + d1 * d1; __syncthreads();
    for (int s2 = 128; s2 > 0; s2 >>= 1) {
        if (tid < s2) red[tid] += red[tid + s2];
        __syncthreads();
    }
    const float var = red[0] * (1.0f / (float)Dc);
    const float rs  = rsqrtf(var + 1e-5f);
    const float y0 = d0 * rs * g[tid]       + be[tid];
    const float y1 = d1 * rs * g[256 + tid] + be[256 + tid];
    h[base + tid] = y0;        h[base + 256 + tid] = y1;
    h16[base + tid] = (_Float16)y0; h16[base + 256 + tid] = (_Float16)y1;
}

// ---------------------------------------------------------------------------
// MoE gate: softmax over E=4 logits, top-2, renormalize -> cw[T,E]
// ---------------------------------------------------------------------------
__global__ void moe_gate(const float* __restrict__ h, const float* __restrict__ gw,
                         const float* __restrict__ gb, float* __restrict__ cw)
{
    const int t = blockIdx.x * 256 + threadIdx.x;
    if (t >= Tc) return;
    const float* hr = h + (long long)t * Dc;
    float lg[Ec];
#pragma unroll
    for (int e = 0; e < Ec; ++e) {
        float s = gb[e];
        const float* wr = gw + (long long)e * Dc;
        for (int d = 0; d < Dc; ++d) s += hr[d] * wr[d];
        lg[e] = s;
    }
    float m = lg[0];
#pragma unroll
    for (int e = 1; e < Ec; ++e) m = fmaxf(m, lg[e]);
    float p[Ec]; float sum = 0.f;
#pragma unroll
    for (int e = 0; e < Ec; ++e) { p[e] = __expf(lg[e] - m); sum += p[e]; }
#pragma unroll
    for (int e = 0; e < Ec; ++e) p[e] /= sum;
    int i1 = 0;
#pragma unroll
    for (int e = 1; e < Ec; ++e) if (p[e] > p[i1]) i1 = e;
    int i2 = (i1 == 0) ? 1 : 0;
#pragma unroll
    for (int e = 0; e < Ec; ++e) if (e != i1 && p[e] > p[i2]) i2 = e;
    const float wsum = p[i1] + p[i2];
#pragma unroll
    for (int e = 0; e < Ec; ++e)
        cw[(long long)t * Ec + e] = (e == i1 || e == i2) ? (p[e] / wsum) : 0.0f;
}

// ---------------------------------------------------------------------------
// moe += cw[:,e] * eo   (init on first expert)
// ---------------------------------------------------------------------------
__global__ void moe_combine(float* __restrict__ moe, const float* __restrict__ eo,
                            const float* __restrict__ cw, int e, int init)
{
    const long long i = (long long)blockIdx.x * 256 + threadIdx.x;   // < T*D
    const int t = (int)(i >> 9);
    const float w = cw[(long long)t * Ec + e];
    const float v = w * eo[i];
    if (init) moe[i] = v; else moe[i] += v;
}

// ---------------------------------------------------------------------------
// Head: out[b,j] = h[b,S-1,:] . hw[j,:] + hb[j]
// ---------------------------------------------------------------------------
__global__ void head_out(const float* __restrict__ h, const float* __restrict__ hw,
                         const float* __restrict__ hb, float* __restrict__ out)
{
    __shared__ float red[256];
    const int b = blockIdx.x / 3, j = blockIdx.x % 3;
    const float* hr = h + ((long long)b * Sc + (Sc - 1)) * Dc;
    float s = 0.f;
    for (int d = threadIdx.x; d < Dc; d += 256) s += hr[d] * hw[j * Dc + d];
    red[threadIdx.x] = s; __syncthreads();
    for (int s2 = 128; s2 > 0; s2 >>= 1) {
        if (threadIdx.x < s2) red[threadIdx.x] += red[threadIdx.x + s2];
        __syncthreads();
    }
    if (threadIdx.x == 0) out[b * 3 + j] = red[0] + hb[j];
}

// ---------------------------------------------------------------------------
// f32 -> f16 convert (weights)
// ---------------------------------------------------------------------------
__global__ void f32_to_f16(const float* __restrict__ s, _Float16* __restrict__ d,
                           long long n)
{
    const long long i = (long long)blockIdx.x * 256 + threadIdx.x;
    if (i < n) d[i] = (_Float16)s[i];
}

// ---------------------------------------------------------------------------
static void launch_gemm(hipStream_t st,
                        const _Float16* A, long long sA,
                        const _Float16* W, long long sW,
                        const float* bias,
                        float* C, int ldC, long long sC,
                        _Float16* C16, int ldC16, long long sC16,
                        int M, int N, int K, int relu, int batch)
{
    dim3 g((N + 127) / 128, (M + 127) / 128, batch);
    gemm_wmma_f16<<<g, 256, 0, st>>>(A, sA, W, sW, bias,
                                     C, ldC, sC, C16, ldC16, sC16,
                                     M, N, K, relu);
}

extern "C" void kernel_launch(void* const* d_in, const int* in_sizes, int n_in,
                              void* d_out, int out_size, void* d_ws, size_t ws_size,
                              hipStream_t stream)
{
    (void)in_sizes; (void)n_in; (void)out_size; (void)ws_size;

    const float* x     = (const float*)d_in[0];
    const float* Wp    = (const float*)d_in[1];
    const float* bp    = (const float*)d_in[2];
    const float* femb  = (const float*)d_in[3];
    const float* qkv_w = (const float*)d_in[4];
    const float* qkv_b = (const float*)d_in[5];
    const float* ow    = (const float*)d_in[6];
    const float* ob    = (const float*)d_in[7];
    const float* g1    = (const float*)d_in[8];
    const float* be1   = (const float*)d_in[9];
    const float* gw    = (const float*)d_in[10];
    const float* gb    = (const float*)d_in[11];
    const float* ew1   = (const float*)d_in[12];
    const float* eb1   = (const float*)d_in[13];
    const float* ew2   = (const float*)d_in[14];
    const float* eb2   = (const float*)d_in[15];
    const float* g2    = (const float*)d_in[16];
    const float* be2   = (const float*)d_in[17];
    const float* hwp   = (const float*)d_in[18];
    const float* hbp   = (const float*)d_in[19];
    const int*   freq  = (const int*)d_in[20];
    float* out = (float*)d_out;

    // ---- workspace carve-out (256B aligned, same layout every call) ----
    char* base = (char*)d_ws;
    size_t off = 0;
    auto alloc = [&](size_t bytes) -> void* {
        off = (off + 255) & ~(size_t)255;
        void* p = base + off;
        off += bytes;
        return p;
    };

    float*     meanB  = (float*)alloc(Bc * Fc * sizeof(float));
    float*     istdB  = (float*)alloc(Bc * Fc * sizeof(float));
    float*     h32    = (float*)alloc((size_t)Tc * Dc * sizeof(float));
    _Float16*  h16    = (_Float16*)alloc((size_t)Tc * Dc * 2);
    _Float16*  qkv16  = (_Float16*)alloc((size_t)Tc * 3 * Dc * 2);
    _Float16*  q16    = (_Float16*)alloc((size_t)Bc * Hc * Sc * HDc * 2);
    _Float16*  k16    = (_Float16*)alloc((size_t)Bc * Hc * Sc * HDc * 2);
    _Float16*  v16t   = (_Float16*)alloc((size_t)Bc * Hc * HDc * Sc * 2);
    float*     sc32   = (float*)alloc((size_t)Hc * Sc * Sc * sizeof(float));
    _Float16*  p16    = (_Float16*)alloc((size_t)Hc * Sc * Sc * 2);
    _Float16*  pv16   = (_Float16*)alloc((size_t)Tc * Dc * 2);
    float*     oproj32= (float*)alloc((size_t)Tc * Dc * sizeof(float));
    float*     cwb    = (float*)alloc((size_t)Tc * Ec * sizeof(float));
    _Float16*  hid16  = (_Float16*)alloc((size_t)Tc * 2 * Dc * 2);
    float*     eo32   = (float*)alloc((size_t)Tc * Dc * sizeof(float));
    float*     moe32  = (float*)alloc((size_t)Tc * Dc * sizeof(float));
    _Float16*  qkvw16 = (_Float16*)alloc((size_t)Lc * 3 * Dc * Dc * 2);
    _Float16*  ow16   = (_Float16*)alloc((size_t)Lc * Dc * Dc * 2);
    _Float16*  ew1w16 = (_Float16*)alloc((size_t)Lc * Ec * 2 * Dc * Dc * 2);
    _Float16*  ew2w16 = (_Float16*)alloc((size_t)Lc * Ec * Dc * 2 * Dc * 2);

    const int TD_BLOCKS = (Tc * Dc) / 256;   // 16384

    // ---- weight conversion to f16 (once per call) ----
    {
        long long n;
        n = (long long)Lc * 3 * Dc * Dc;
        f32_to_f16<<<(unsigned)((n + 255) / 256), 256, 0, stream>>>(qkv_w, qkvw16, n);
        n = (long long)Lc * Dc * Dc;
        f32_to_f16<<<(unsigned)((n + 255) / 256), 256, 0, stream>>>(ow, ow16, n);
        n = (long long)Lc * Ec * 2 * Dc * Dc;
        f32_to_f16<<<(unsigned)((n + 255) / 256), 256, 0, stream>>>(ew1, ew1w16, n);
        n = (long long)Lc * Ec * Dc * 2 * Dc;
        f32_to_f16<<<(unsigned)((n + 255) / 256), 256, 0, stream>>>(ew2, ew2w16, n);
    }

    // ---- instance norm + projection + embeddings ----
    inst_stats<<<Bc * Fc, 256, 0, stream>>>(x, meanB, istdB);
    proj_embed<<<TD_BLOCKS, 256, 0, stream>>>(x, meanB, istdB, Wp, bp, femb, freq,
                                              h32, h16);

    // ---- transformer layers ----
    for (int l = 0; l < Lc; ++l) {
        // QKV projection  [T,512] x [1536,512]^T -> f16 [T,1536]
        launch_gemm(stream, h16, 0,
                    qkvw16 + (size_t)l * 3 * Dc * Dc, 0,
                    qkv_b + (size_t)l * 3 * Dc,
                    nullptr, 0, 0, qkv16, 3 * Dc, 0,
                    Tc, 3 * Dc, Dc, 0, 1);
        split_qkv<<<TD_BLOCKS, 256, 0, stream>>>(qkv16, q16, k16, v16t);

        // attention, batched over heads, looped over batch b
        for (int b = 0; b < Bc; ++b) {
            const long long qoff = (long long)b * Hc * Sc * HDc;
            // scores = Q K^T  (M=S, N=S, K=HD)
            launch_gemm(stream, q16 + qoff, (long long)Sc * HDc,
                        k16 + qoff, (long long)Sc * HDc, nullptr,
                        sc32, Sc, (long long)Sc * Sc,
                        nullptr, 0, 0,
                        Sc, Sc, HDc, 0, Hc);
            softmax_rows<<<dim3(Sc, 1, Hc), 256, 0, stream>>>(
                sc32, (long long)Sc * Sc, p16, (long long)Sc * Sc, 0.125f);
            // O = P V, heads written interleaved into pv16 [T, D]
            launch_gemm(stream, p16, (long long)Sc * Sc,
                        v16t + (long long)b * Hc * HDc * Sc, (long long)HDc * Sc,
                        nullptr,
                        nullptr, 0, 0,
                        pv16 + (long long)b * Sc * Dc, Dc, (long long)HDc,
                        Sc, HDc, Sc, 0, Hc);
        }

        // output projection + residual LN
        launch_gemm(stream, pv16, 0,
                    ow16 + (size_t)l * Dc * Dc, 0, ob + (size_t)l * Dc,
                    oproj32, Dc, 0, nullptr, 0, 0,
                    Tc, Dc, Dc, 0, 1);
        add_ln<<<Tc, 256, 0, stream>>>(h32, oproj32,
                                       g1 + (size_t)l * Dc, be1 + (size_t)l * Dc, h16);

        // MoE: top-2 gate, loop experts densely (matches reference math)
        moe_gate<<<(Tc + 255) / 256, 256, 0, stream>>>(
            h32, gw + (size_t)l * Ec * Dc, gb + (size_t)l * Ec, cwb);
        for (int e = 0; e < Ec; ++e) {
            const size_t wo1 = ((size_t)l * Ec + e) * 2 * Dc * Dc;
            const size_t wo2 = ((size_t)l * Ec + e) * Dc * 2 * Dc;
            launch_gemm(stream, h16, 0,
                        ew1w16 + wo1, 0, eb1 + ((size_t)l * Ec + e) * 2 * Dc,
                        nullptr, 0, 0, hid16, 2 * Dc, 0,
                        Tc, 2 * Dc, Dc, 1, 1);                 // ReLU
            launch_gemm(stream, hid16, 0,
                        ew2w16 + wo2, 0, eb2 + ((size_t)l * Ec + e) * Dc,
                        eo32, Dc, 0, nullptr, 0, 0,
                        Tc, Dc, 2 * Dc, 0, 1);
            moe_combine<<<TD_BLOCKS, 256, 0, stream>>>(moe32, eo32, cwb, e,
                                                       (e == 0) ? 1 : 0);
        }
        add_ln<<<Tc, 256, 0, stream>>>(h32, moe32,
                                       g2 + (size_t)l * Dc, be2 + (size_t)l * Dc, h16);
    }

    // ---- head: quantile outputs [B,3] ----
    head_out<<<Bc * 3, 256, 0, stream>>>(h32, hwp, hbp, out);
}
